// NonLocalBlock_34170759807013
// MI455X (gfx1250) — matrix operand order repaired and checked by reading, not compiled, not run on hardware
//
#include <hip/hip_runtime.h>
#include <hip/hip_bf16.h>
#include <stdint.h>

#define BDIM 256              // 8 waves
#define B_   16
#define C_   512
#define CH_  256
#define N_   2048
#define NTOT (B_ * C_ * N_)   // 16,777,216 elements

typedef __attribute__((ext_vector_type(16))) __bf16   v16bf;
typedef __attribute__((ext_vector_type(8)))  float    v8f;
typedef __attribute__((ext_vector_type(8)))  uint32_t v8u;

// ---- helpers -------------------------------------------------------------

__device__ __forceinline__ uint16_t f2bf(float f) {
  uint32_t u = __builtin_bit_cast(uint32_t, f);
  u += 0x7FFFu + ((u >> 16) & 1u);          // round-to-nearest-even
  return (uint16_t)(u >> 16);
}

// A fragment: 16x32 bf16. LDS layout sA[row][k2] (16 dwords per row).
// lane<16: M=lane,    K dwords {0..3, 8..11}
// lane>=16: M=lane-16, K dwords {4..7, 12..15}
__device__ __forceinline__ v16bf frag_a(const uint32_t* sA, int mRow, int lane) {
  const int lsub = lane & 15, hi = lane >> 4;
  const uint32_t* rp = sA + (mRow + lsub) * 16 + hi * 4;
  v8u r;
#pragma unroll
  for (int j = 0; j < 4; ++j) r[j] = rp[j];
#pragma unroll
  for (int j = 0; j < 4; ++j) r[4 + j] = rp[8 + j];
  return __builtin_bit_cast(v16bf, r);
}

// B fragment: 32x16 bf16, staged transposed: sB[n][k2].
// lane<16: N=lane, K dwords {0..3, 8..11}; lane>=16: N=lane-16, K dwords {4..7, 12..15}.
__device__ __forceinline__ v16bf frag_b(const uint32_t* sB, int nCol, int lane) {
  const int lsub = lane & 15, hi = lane >> 4;
  const uint32_t* cp = sB + (nCol + lsub) * 16 + hi * 4;
  v8u r;
#pragma unroll
  for (int j = 0; j < 4; ++j) r[j] = cp[j];
#pragma unroll
  for (int j = 0; j < 4; ++j) r[4 + j] = cp[8 + j];
  return __builtin_bit_cast(v16bf, r);
}

__device__ __forceinline__ v8f wmma_bf16(v16bf a, v16bf b, v8f c) {
  return __builtin_amdgcn_wmma_f32_16x16x32_bf16(false, a, false, b, (short)0, c,
                                                 false, false);
}

// Async global->LDS copy of a 128-row x 64-byte tile (8 KiB, 512 b128 chunks).
// Rows are 64B contiguous in global, strided by rowStrideBytes.
__device__ __forceinline__ void async_tile(uint32_t ldsBase, const uint16_t* g,
                                           uint32_t rowStrideBytes, int tid) {
  uint64_t base = (uint64_t)(uintptr_t)g;
#pragma unroll
  for (int it = 0; it < 2; ++it) {
    uint32_t chunk = (uint32_t)(it * BDIM + tid);          // 0..511
    uint32_t row = chunk >> 2, col = (chunk & 3u) * 16u;
    uint32_t lds = ldsBase + chunk * 16u;
    uint32_t voff = row * rowStrideBytes + col;
    asm volatile("global_load_async_to_lds_b128 %0, %1, %2"
                 : : "v"(lds), "v"(voff), "s"(base) : "memory");
  }
}

// Shared 128x128-tile GEMM body: C[128,128] += A[128,K] * B_rows[128,K]^T
// (B staged as rows of length K, i.e. LDS holds B^T). 8 waves: waveM in 0..3
// covers 32 rows, waveN in 0..1 covers 64 cols.
__device__ __forceinline__ void gemm_tile(const uint16_t* Abase, int lda,
                                          const uint16_t* Bbase, int ldb, int K,
                                          uint16_t* sA16, uint16_t* sB16,
                                          v8f acc[2][4]) {
  const int tid = threadIdx.x, lane = tid & 31, wave = tid >> 5;
  const int waveM = wave & 3, waveN = wave >> 2;
  const uint32_t ldsA = (uint32_t)(uintptr_t)sA16;
  const uint32_t ldsB = (uint32_t)(uintptr_t)sB16;
  const uint32_t* sA = (const uint32_t*)sA16;
  const uint32_t* sB = (const uint32_t*)sB16;

  for (int k0 = 0; k0 < K; k0 += 32) {
    async_tile(ldsA, Abase + k0, (uint32_t)(lda * 2), tid);
    async_tile(ldsB, Bbase + k0, (uint32_t)(ldb * 2), tid);
    asm volatile("s_wait_asynccnt 0x0" ::: "memory");
    __syncthreads();
    v16bf a0 = frag_a(sA, waveM * 32, lane);
    v16bf a1 = frag_a(sA, waveM * 32 + 16, lane);
#pragma unroll
    for (int s = 0; s < 4; ++s) {
      v16bf bb = frag_b(sB, waveN * 64 + s * 16, lane);
      acc[0][s] = wmma_bf16(a0, bb, acc[0][s]);
      acc[1][s] = wmma_bf16(a1, bb, acc[1][s]);
    }
    __syncthreads();
  }
}

// ---- kernel 0a: convert weights to bf16 (concatenated theta/phi/g + Ww) --
__global__ __launch_bounds__(256) void k_cvt_w(
    const float* __restrict__ thw, const float* __restrict__ phw,
    const float* __restrict__ gw,  const float* __restrict__ Ww,
    const float* __restrict__ thb, const float* __restrict__ phb,
    const float* __restrict__ gb,
    uint16_t* __restrict__ Wcat, uint16_t* __restrict__ Ww16,
    float* __restrict__ bcat) {
  const int WN = CH_ * C_;  // 131072
  int i = blockIdx.x * 256 + threadIdx.x;
  if (i < WN)             Wcat[i] = f2bf(thw[i]);
  else if (i < 2 * WN)    Wcat[i] = f2bf(phw[i - WN]);
  else if (i < 3 * WN)    Wcat[i] = f2bf(gw[i - 2 * WN]);
  else if (i < 4 * WN)    Ww16[i - 3 * WN] = f2bf(Ww[i - 3 * WN]);
  if (i < CH_) { bcat[i] = thb[i]; bcat[CH_ + i] = phb[i]; bcat[2 * CH_ + i] = gb[i]; }
}

// ---- kernel 0b: x -> bf16, transposed per batch to Xt[b][sp][ch] ---------
// grid (N/64, C/64, B), block 256, LDS-tiled transpose
__global__ __launch_bounds__(256) void k_cvt_x(const float* __restrict__ x,
                                               uint16_t* __restrict__ Xt) {
  __shared__ uint16_t t[64][65];
  const int tid = threadIdx.x, b = blockIdx.z;
  const int n0 = blockIdx.x * 64, c0 = blockIdx.y * 64;
  const float* xb = x + (size_t)b * C_ * N_;
  uint16_t* xtb = Xt + (size_t)b * N_ * C_;
  for (int idx = tid; idx < 64 * 64; idx += 256) {
    int cc = idx >> 6, nn = idx & 63;                     // coalesced in nn
    t[cc][nn] = f2bf(xb[(size_t)(c0 + cc) * N_ + n0 + nn]);
  }
  __syncthreads();
  for (int idx = tid; idx < 64 * 64; idx += 256) {
    int nn = idx >> 6, cc = idx & 63;                     // coalesced in cc
    xtb[(size_t)(n0 + nn) * C_ + c0 + cc] = t[cc][nn];
  }
}

// ---- kernel 1: T/P/G projections ([768,512] x [512,2048] per batch) ------
// grid (N/128=16, 768/128=6, B). theta output stored transposed (Tt[n][ch]).
__global__ __launch_bounds__(BDIM) void k_proj(
    const uint16_t* __restrict__ Wcat, const float* __restrict__ bcat,
    const uint16_t* __restrict__ Xt,
    uint16_t* __restrict__ Tt, uint16_t* __restrict__ P,
    uint16_t* __restrict__ G) {
  __shared__ __align__(16) uint16_t sA16[128 * 32];
  __shared__ __align__(16) uint16_t sB16[128 * 32];
  const int lane = threadIdx.x & 31, wave = threadIdx.x >> 5;
  const int waveM = wave & 3, waveN = wave >> 2;
  const int b = blockIdx.z, m0 = blockIdx.y * 128, n0 = blockIdx.x * 128;

  v8f acc[2][4] = {};
  // A rows: Wcat[m0..m0+127][0..511]; B rows: Xt[n0..n0+127][0..511]
  gemm_tile(Wcat + (size_t)m0 * C_, C_,
            Xt + ((size_t)b * N_ + n0) * C_, C_, C_, sA16, sB16, acc);

  const int lsub = lane & 15, hi = lane >> 4;
  const int proj = m0 >> 8;                 // uniform per block
#pragma unroll
  for (int r = 0; r < 2; ++r)
#pragma unroll
    for (int s = 0; s < 4; ++s) {
      int ng = n0 + waveN * 64 + s * 16 + lsub;
#pragma unroll
      for (int i = 0; i < 8; ++i) {
        int mg = m0 + waveM * 32 + r * 16 + hi * 8 + i;     // 0..767
        int ml = mg & (CH_ - 1);
        uint16_t bv = f2bf(acc[r][s][i] + bcat[mg]);
        if (proj == 0)      Tt[((size_t)b * N_ + ng) * CH_ + ml] = bv;
        else if (proj == 1) P[((size_t)b * CH_ + ml) * N_ + ng] = bv;
        else                G[((size_t)b * CH_ + ml) * N_ + ng] = bv;
      }
    }
}

// ---- kernel 2: S = P * G^T / N -> bf16 [256,256] per batch ---------------
// grid (2, 2, B)
__global__ __launch_bounds__(BDIM) void k_affinity(
    const uint16_t* __restrict__ P, const uint16_t* __restrict__ G,
    uint16_t* __restrict__ S16) {
  __shared__ __align__(16) uint16_t sA16[128 * 32];
  __shared__ __align__(16) uint16_t sB16[128 * 32];
  const int lane = threadIdx.x & 31, wave = threadIdx.x >> 5;
  const int waveM = wave & 3, waveN = wave >> 2;
  const int b = blockIdx.z, m0 = blockIdx.y * 128, n0 = blockIdx.x * 128;

  v8f acc[2][4] = {};
  gemm_tile(P + ((size_t)b * CH_ + m0) * N_, N_,
            G + ((size_t)b * CH_ + n0) * N_, N_, N_, sA16, sB16, acc);

  const int lsub = lane & 15, hi = lane >> 4;
  uint16_t* Sb = S16 + (size_t)b * CH_ * CH_;
  const float inv = 1.0f / (float)N_;
#pragma unroll
  for (int r = 0; r < 2; ++r)
#pragma unroll
    for (int s = 0; s < 4; ++s) {
      int ng = n0 + waveN * 64 + s * 16 + lsub;
#pragma unroll
      for (int i = 0; i < 8; ++i) {
        int mg = m0 + waveM * 32 + r * 16 + hi * 8 + i;
        Sb[mg * CH_ + ng] = f2bf(acc[r][s][i] * inv);
      }
    }
}

// ---- kernel 3: M_b = W_w * S_b^T  ([512,256] x [256,256]) -> bf16 --------
// grid (2, 4, B)
__global__ __launch_bounds__(BDIM) void k_fold(
    const uint16_t* __restrict__ Ww16, const uint16_t* __restrict__ S16,
    uint16_t* __restrict__ Mmat) {
  __shared__ __align__(16) uint16_t sA16[128 * 32];
  __shared__ __align__(16) uint16_t sB16[128 * 32];
  const int lane = threadIdx.x & 31, wave = threadIdx.x >> 5;
  const int waveM = wave & 3, waveN = wave >> 2;
  const int b = blockIdx.z, m0 = blockIdx.y * 128, n0 = blockIdx.x * 128;

  v8f acc[2][4] = {};
  gemm_tile(Ww16 + (size_t)m0 * CH_, CH_,
            S16 + ((size_t)b * CH_ + n0) * CH_, CH_, CH_, sA16, sB16, acc);

  const int lsub = lane & 15, hi = lane >> 4;
  uint16_t* Mb = Mmat + (size_t)b * C_ * CH_;
#pragma unroll
  for (int r = 0; r < 2; ++r)
#pragma unroll
    for (int s = 0; s < 4; ++s) {
      int ng = n0 + waveN * 64 + s * 16 + lsub;
#pragma unroll
      for (int i = 0; i < 8; ++i) {
        int mg = m0 + waveM * 32 + r * 16 + hi * 8 + i;
        Mb[mg * CH_ + ng] = f2bf(acc[r][s][i]);
      }
    }
}

// ---- kernel 4: w = M_b * Tt^T + W_b -> d_out (fp32) ----------------------
// grid (16, 4, B)
__global__ __launch_bounds__(BDIM) void k_wconv(
    const uint16_t* __restrict__ Mmat, const uint16_t* __restrict__ Tt,
    const float* __restrict__ Wb, float* __restrict__ w) {
  __shared__ __align__(16) uint16_t sA16[128 * 32];
  __shared__ __align__(16) uint16_t sB16[128 * 32];
  const int lane = threadIdx.x & 31, wave = threadIdx.x >> 5;
  const int waveM = wave & 3, waveN = wave >> 2;
  const int b = blockIdx.z, m0 = blockIdx.y * 128, n0 = blockIdx.x * 128;

  v8f acc[2][4] = {};
  gemm_tile(Mmat + ((size_t)b * C_ + m0) * CH_, CH_,
            Tt + ((size_t)b * N_ + n0) * CH_, CH_, CH_, sA16, sB16, acc);

  const int lsub = lane & 15, hi = lane >> 4;
  float* wb_out = w + (size_t)b * C_ * N_;
#pragma unroll
  for (int r = 0; r < 2; ++r)
#pragma unroll
    for (int s = 0; s < 4; ++s) {
      int ng = n0 + waveN * 64 + s * 16 + lsub;
#pragma unroll
      for (int i = 0; i < 8; ++i) {
        int mg = m0 + waveM * 32 + r * 16 + hi * 8 + i;
        wb_out[(size_t)mg * N_ + ng] = acc[r][s][i] + Wb[mg];
      }
    }
}

// ---- kernel 5: per-channel sums over (B, N) ------------------------------
__global__ __launch_bounds__(256) void k_stats(const float* __restrict__ w,
                                               float* __restrict__ sums,
                                               float* __restrict__ sqs) {
  __shared__ float sh[256], sh2[256];
  const int c = blockIdx.x, tid = threadIdx.x;
  float s = 0.f, q = 0.f;
  for (int b = 0; b < B_; ++b) {
    const float* row = w + ((size_t)b * C_ + c) * N_;
    for (int n = tid; n < N_; n += 256) { float v = row[n]; s += v; q += v * v; }
  }
  sh[tid] = s; sh2[tid] = q;
  __syncthreads();
  for (int off = 128; off > 0; off >>= 1) {
    if (tid < off) { sh[tid] += sh[tid + off]; sh2[tid] += sh2[tid + off]; }
    __syncthreads();
  }
  if (tid == 0) { sums[c] = sh[0]; sqs[c] = sh2[0]; }
}

// ---- kernel 6: scale/shift -----------------------------------------------
__global__ void k_finalize(const float* __restrict__ sums,
                           const float* __restrict__ sqs,
                           const float* __restrict__ gamma,
                           const float* __restrict__ beta,
                           float* __restrict__ scale, float* __restrict__ shift) {
  int c = threadIdx.x;               // 512 threads
  const float cnt = (float)(B_ * N_);
  float mean = sums[c] / cnt;
  float var  = sqs[c] / cnt - mean * mean;
  float sc   = gamma[c] * rsqrtf(var + 1e-5f);
  scale[c] = sc;
  shift[c] = beta[c] - mean * sc;
}

// ---- kernel 7: BN apply + residual, in-place on d_out --------------------
__global__ __launch_bounds__(256) void k_out(float* __restrict__ out,
                                             const float* __restrict__ x,
                                             const float* __restrict__ scale,
                                             const float* __restrict__ shift) {
  size_t e = ((size_t)blockIdx.x * 256 + threadIdx.x) * 4;
  if (e >= (size_t)NTOT) return;
  int c = (int)((e >> 11) & (C_ - 1));     // N_=2048 per channel row
  float4 wv = *(const float4*)(out + e);
  float4 xv = *(const float4*)(x + e);
  float sc = scale[c], sh = shift[c];
  float4 r;
  r.x = wv.x * sc + sh + xv.x;
  r.y = wv.y * sc + sh + xv.y;
  r.z = wv.z * sc + sh + xv.z;
  r.w = wv.w * sc + sh + xv.w;
  *(float4*)(out + e) = r;
}

// ---- launch --------------------------------------------------------------
extern "C" void kernel_launch(void* const* d_in, const int* in_sizes, int n_in,
                              void* d_out, int out_size, void* d_ws, size_t ws_size,
                              hipStream_t stream) {
  (void)in_sizes; (void)n_in; (void)out_size; (void)ws_size;
  const float* x   = (const float*)d_in[0];
  const float* thw = (const float*)d_in[1];
  const float* thb = (const float*)d_in[2];
  const float* phw = (const float*)d_in[3];
  const float* phb = (const float*)d_in[4];
  const float* gw  = (const float*)d_in[5];
  const float* gb  = (const float*)d_in[6];
  const float* Ww  = (const float*)d_in[7];
  const float* Wb  = (const float*)d_in[8];
  const float* gam = (const float*)d_in[9];
  const float* bet = (const float*)d_in[10];
  float* out = (float*)d_out;

  // workspace layout (bytes)
  char* ws = (char*)d_ws;
  uint16_t* Xt    = (uint16_t*)(ws);                         // 33,554,432
  uint16_t* Tt    = (uint16_t*)(ws + 33554432);              // 16,777,216
  uint16_t* P     = (uint16_t*)(ws + 50331648);              // 16,777,216
  uint16_t* G     = (uint16_t*)(ws + 67108864);              // 16,777,216
  uint16_t* S16   = (uint16_t*)(ws + 83886080);              //  2,097,152
  uint16_t* Mmat  = (uint16_t*)(ws + 85983232);              //  4,194,304
  uint16_t* Wcat  = (uint16_t*)(ws + 90177536);              //    786,432
  uint16_t* Ww16  = (uint16_t*)(ws + 90963968);              //    262,144
  float*    bcat  = (float*)   (ws + 91226112);              //      3,072
  float*    sums  = (float*)   (ws + 91229184);
  float*    sqs   = sums + C_;
  float*    scale = sums + 2 * C_;
  float*    shift = sums + 3 * C_;

  k_cvt_w   <<<(4 * CH_ * C_) / 256, 256, 0, stream>>>(
      thw, phw, gw, Ww, thb, phb, gb, Wcat, Ww16, bcat);
  k_cvt_x   <<<dim3(N_ / 64, C_ / 64, B_), 256, 0, stream>>>(x, Xt);
  k_proj    <<<dim3(N_ / 128, (3 * CH_) / 128, B_), BDIM, 0, stream>>>(
      Wcat, bcat, Xt, Tt, P, G);
  k_affinity<<<dim3(CH_ / 128, CH_ / 128, B_), BDIM, 0, stream>>>(P, G, S16);
  k_fold    <<<dim3(CH_ / 128, C_ / 128, B_), BDIM, 0, stream>>>(Ww16, S16, Mmat);
  k_wconv   <<<dim3(N_ / 128, C_ / 128, B_), BDIM, 0, stream>>>(Mmat, Tt, Wb, out);
  k_stats   <<<C_, 256, 0, stream>>>(out, sums, sqs);
  k_finalize<<<1, C_, 0, stream>>>(sums, sqs, gam, bet, scale, shift);
  k_out     <<<(NTOT / 4 + 255) / 256, 256, 0, stream>>>(out, x, scale, shift);
}
